// STDPHopfieldModel_76089640616056
// MI455X (gfx1250) — matrix operand order repaired and checked by reading, not compiled, not run on hardware
//
#include <hip/hip_runtime.h>

typedef __attribute__((ext_vector_type(16))) _Float16 v16h;
typedef __attribute__((ext_vector_type(8)))  float    v8f;
typedef __attribute__((vector_size(8)))      int      v2i;

#define DECAY 0.95122945f   // exp(-DT/TAU) = exp(-0.05), fp32
#define VTH   0.5f

#ifndef __has_builtin
#define __has_builtin(x) 0
#endif
#if __has_builtin(__builtin_amdgcn_global_load_async_to_lds_b64)
#define HAVE_ASYNC_LDS 1
#else
#define HAVE_ASYNC_LDS 0
#endif
#define AS1 __attribute__((address_space(1)))
#define AS3 __attribute__((address_space(3)))

// ---------------------------------------------------------------------------
// Kernel 0: pack w2 (f32, OIHW 128x64x5x5) into WMMA B-fragment layout (f16).
// Bpack[nt(8)][kc(50)][lane(32)][j(16)], K mapping k = tap*64 + ci,
// B-layout: lane half h, element j holds chunk-local K (j>>3)*16 + 8h + (j&7).
// ---------------------------------------------------------------------------
__global__ void __launch_bounds__(256) pack_w2(const float* __restrict__ w2,
                                               _Float16* __restrict__ bp) {
    int idx = blockIdx.x * 256 + threadIdx.x;
    if (idx >= 8 * 50 * 32 * 16) return;
    int j    = idx & 15;
    int lane = (idx >> 4) & 31;
    int kc   = (idx >> 9) % 50;
    int nt   = idx / (50 * 32 * 16);
    int k    = kc * 32 + ((j >> 3) << 4) + ((lane >> 4) << 3) + (j & 7);   // 0..1599
    int tap  = k >> 6;                             // 0..24
    int ci   = k & 63;                             // 0..63
    int co   = nt * 16 + (lane & 15);              // 0..127
    bp[idx] = (_Float16)w2[(co * 64 + ci) * 25 + tap];
}

// ---------------------------------------------------------------------------
// Kernel 1: conv1 + LIF1 (WTA over 64 ch) + 2x2 pool, 16 timesteps fused.
// One block per image, one thread per 32x32 pixel; the pixel's 64 membrane
// values live in registers so the WTA argmax is thread-local. Emits per-step
// 16x16 quad spike bitmasks (64 bits/quad -> 512 u32 per image per step).
// ---------------------------------------------------------------------------
__global__ void __launch_bounds__(1024) lif1_kernel(
        const float* __restrict__ sup, const float* __restrict__ qry,
        const float* __restrict__ w1, unsigned* __restrict__ p1bits) {
    int blk = blockIdx.x;
    int set = blk >> 7, img = blk & 127;
    const float* sp = (set == 0) ? sup : qry;

    __shared__ float    w1f[1600];
    __shared__ unsigned srow[32];
    __shared__ unsigned q1[512];

    int tid = threadIdx.x;
    w1f[tid & 1023] = w1[tid & 1023];
    if (tid < 576) w1f[1024 + tid] = w1[1024 + tid];
    int y = tid >> 5, x = tid & 31;

    float mem[64];
#pragma unroll
    for (int c = 0; c < 64; c++) mem[c] = 0.f;

    for (int t = 0; t < 16; t++) {
        if (tid < 32)  srow[tid] = 0u;
        if (tid < 512) q1[tid]   = 0u;
        __syncthreads();
        float sv = sp[((t * 128 + img) * 32 + y) * 32 + x];
        if (sv >= 0.5f) atomicOr(&srow[y], 1u << x);
        __syncthreads();

        float m[25];
#pragma unroll
        for (int tap = 0; tap < 25; tap++) {
            int dy = tap / 5 - 2, dx = tap % 5 - 2;
            int iy = y + dy, ix = x + dx;
            unsigned bit = (iy >= 0 && iy < 32 && ix >= 0 && ix < 32)
                               ? ((srow[iy] >> ix) & 1u) : 0u;
            m[tap] = (float)bit;
        }
        float best = -1e30f; int arg = 0;
#pragma unroll
        for (int c = 0; c < 64; c++) {
            float I = 0.f;
#pragma unroll
            for (int tap = 0; tap < 25; tap++) I += m[tap] * w1f[c * 25 + tap];
            float nm = DECAY * mem[c] + I;
            mem[c] = nm;
            if (nm > best) { best = nm; arg = c; }
        }
        int spike = (best >= VTH);
#pragma unroll
        for (int c = 0; c < 64; c++)
            mem[c] = (spike && c == arg) ? 0.f : mem[c];
        if (spike) {
            int q = (y >> 1) * 16 + (x >> 1);
            atomicOr(&q1[q * 2 + (arg >> 5)], 1u << (arg & 31));
        }
        __syncthreads();
        if (tid < 512)
            p1bits[((set * 128 + img) * 16 + t) * 512 + tid] = q1[tid];
        __syncthreads();
    }
}

// ---------------------------------------------------------------------------
// Kernel 2: conv2 as implicit GEMM on v_wmma_f32_16x16x32_f16, fused with
// LIF2 (WTA over 128 ch), 2x2 pool and spike-count accumulation across the
// 16 sequential timesteps. One block (256 thr = 8 waves) per image.
//   Per step: M=256 pixels x N=128 ch x K=1600 (64 ci x 25 taps).
//   Wave w owns N-tile w; kc-outer / mt-unrolled keeps 16 v8f accumulators
//   in VGPRs (B fragment loaded once per K-chunk). A fragments expand from
//   a zero-padded 20x20 spike-bitmask halo (branchless ds_load_b64) through
//   a 256-entry byte->8xf16 LUT (2x ds_load_b128 per fragment).
// Dynamic LDS: LUT 4K + halo 3.2K + I2 128K + mem2 128K + qmask 1K +
//              counts 32K = 296.1K of the WGP's 320K.
// ---------------------------------------------------------------------------
#define SM_LUT   0
#define SM_P1H   4096
#define SM_I2    7296
#define SM_MEM2  138368
#define SM_QM    269440
#define SM_CNT   270464
#define SM_TOTAL 303232

__global__ void __launch_bounds__(256) conv2_lif2_kernel(
        const unsigned* __restrict__ p1bits, const _Float16* __restrict__ bp,
        float* __restrict__ rate) {
    extern __shared__ char smem[];
    uint4*    lut  = (uint4*)(smem + SM_LUT);    // 256 x 16B byte->8 halves
    uint2*    p1h  = (uint2*)(smem + SM_P1H);    // 20x20 halo of 64b masks
    float*    I2   = (float*)(smem + SM_I2);     // [ch][pix] 128x256
    float*    mem2 = (float*)(smem + SM_MEM2);   // [ch][pix] 128x256
    unsigned* qm   = (unsigned*)(smem + SM_QM);  // 64 quads x 4 words
    float*    cnt  = (float*)(smem + SM_CNT);    // [ch][quad] 128x64

    int tid = threadIdx.x, gi = blockIdx.x;      // gi = set*128 + img
    int wv  = tid >> 5, lane = tid & 31;
    int laneoff = (lane & 16) ? 8 : 0;           // A-frag K sub-offset
    int xr  = lane & 15;

    // ---- one-time init: LUT, halo borders, membranes, counts ----
    {
        unsigned b = (unsigned)tid;              // 0..255
        uint4 e;
        e.x = (((b >> 0) & 1) ? 0x3C00u : 0u) | (((b >> 1) & 1) ? 0x3C000000u : 0u);
        e.y = (((b >> 2) & 1) ? 0x3C00u : 0u) | (((b >> 3) & 1) ? 0x3C000000u : 0u);
        e.z = (((b >> 4) & 1) ? 0x3C00u : 0u) | (((b >> 5) & 1) ? 0x3C000000u : 0u);
        e.w = (((b >> 6) & 1) ? 0x3C00u : 0u) | (((b >> 7) & 1) ? 0x3C000000u : 0u);
        lut[b] = e;
    }
    p1h[tid] = make_uint2(0u, 0u);
    if (tid < 144) p1h[256 + tid] = make_uint2(0u, 0u);
    for (int i = tid; i < 32768; i += 256) mem2[i] = 0.f;
    for (int i = tid; i < 8192;  i += 256) cnt[i]  = 0.f;
    __syncthreads();

    for (int t = 0; t < 16; t++) {
        // ---- stage spike bitmasks into halo interior (async DMA path) ----
        {
            int qy = tid >> 4, qx = tid & 15;
            uint2* dst = &p1h[(qy + 2) * 20 + (qx + 2)];
            const unsigned* src = p1bits + (gi * 16 + t) * 512 + tid * 2;
#if HAVE_ASYNC_LDS
            __builtin_amdgcn_global_load_async_to_lds_b64(
                (AS1 v2i*)src, (AS3 v2i*)dst, 0, 0);
#if __has_builtin(__builtin_amdgcn_s_wait_asynccnt)
            __builtin_amdgcn_s_wait_asynccnt(0);
#else
            asm volatile("s_wait_asynccnt 0" ::: "memory");
#endif
#else
            *dst = *(const uint2*)src;
#endif
            qm[tid] = 0u;
        }
        __syncthreads();

        // ---- WMMA implicit GEMM: wave wv -> channels [wv*16, wv*16+16) ----
        v8f acc[16];
        {
            v8f vz = {};
#pragma unroll
            for (int i = 0; i < 16; i++) acc[i] = vz;
        }
        const _Float16* bptr = bp + ((wv * 50) * 32 + lane) * 16;
        const char* hbase = (const char*)p1h + xr * 8;
        for (int dyi = 0; dyi < 5; dyi++) {
            for (int dxi = 0; dxi < 5; dxi++) {
                v16h b0 = *(const v16h*)(bptr);
                v16h b1 = *(const v16h*)(bptr + 512);
                bptr += 1024;
                __builtin_prefetch(bptr, 0, 1);
                const char* pb = hbase + (dyi * 160 + dxi * 8);
#pragma unroll
                for (int mt = 0; mt < 16; mt++) {
                    // pixel (y=mt, x=xr); neighbor words, branchless via halo
                    uint2 Wp = *(const uint2*)(pb + mt * 160);
                    unsigned w0 = Wp.x >> laneoff;   // ci 0..31  (even kc)
                    unsigned w1 = Wp.y >> laneoff;   // ci 32..63 (odd kc)
                    union { uint4 q[2]; v16h h; } A0, A1;
                    A0.q[0] = lut[w0 & 0xFF];
                    A0.q[1] = lut[(w0 >> 16) & 0xFF];
                    A1.q[0] = lut[w1 & 0xFF];
                    A1.q[1] = lut[(w1 >> 16) & 0xFF];
                    acc[mt] = __builtin_amdgcn_wmma_f32_16x16x32_f16(
                        false, A0.h, false, b0, (short)0, acc[mt], false, false);
                    acc[mt] = __builtin_amdgcn_wmma_f32_16x16x32_f16(
                        false, A1.h, false, b1, (short)0, acc[mt], false, false);
                }
            }
        }
        // D layout: row m = (lane/16)*8 + r, col n = lane%16
#pragma unroll
        for (int mt = 0; mt < 16; mt++) {
#pragma unroll
            for (int r = 0; r < 8; r++) {
                int m = ((lane >> 4) << 3) + r;
                I2[(wv * 16 + xr) * 256 + (mt * 16 + m)] = acc[mt][r];
            }
        }
        __syncthreads();

        // ---- LIF2 + k-WTA over 128 channels (thread tid owns pixel tid) ----
        {
            int p = tid;
            float best = -1e30f; int arg = 0;
            for (int c = 0; c < 128; c++) {
                float nm = DECAY * mem2[c * 256 + p] + I2[c * 256 + p];
                mem2[c * 256 + p] = nm;
                if (nm > best) { best = nm; arg = c; }
            }
            if (best >= VTH) {
                mem2[arg * 256 + p] = 0.f;
                int q = ((p >> 5) & 7) * 8 + ((p & 15) >> 1); // (y/2)*8+(x/2)
                atomicOr(&qm[q * 4 + (arg >> 5)], 1u << (arg & 31));
            }
        }
        __syncthreads();

        // ---- 2x2 max-pool (quad OR of one-hot spikes) -> count accumulate --
        {
            int q = tid >> 2, wd = tid & 3;
            unsigned bits = qm[q * 4 + wd];
            while (bits) {
                int b = __builtin_ctz(bits);
                bits &= bits - 1;
                cnt[(wd * 32 + b) * 64 + q] += 1.f;
            }
        }
        __syncthreads();
    }
    for (int i = tid; i < 8192; i += 256)
        rate[gi * 8192 + i] = cnt[i] * (1.f / 16.f);
}

// ---------------------------------------------------------------------------
// Kernel 3: projection  emb[i][e] = rate[i][:] . proj_w[e][:]  (256x256x8192)
// ---------------------------------------------------------------------------
__global__ void __launch_bounds__(256) project_kernel(
        const float* __restrict__ rate, const float* __restrict__ pw,
        float* __restrict__ emb) {
    __shared__ float r[8192];
    int i = blockIdx.x, tid = threadIdx.x;
    for (int k = tid; k < 8192; k += 256) r[k] = rate[i * 8192 + k];
    __syncthreads();
    const float* p = pw + tid * 8192;
    float s = 0.f;
    for (int k = 0; k < 8192; k++) s += r[k] * p[k];
    emb[i * 256 + tid] = s;
}

// Kernel 3b: per-row L2 norms (clamped at 1e-12)
__global__ void __launch_bounds__(256) rownorm_kernel(
        const float* __restrict__ emb, float* __restrict__ norms) {
    __shared__ float red[256];
    int i = blockIdx.x, tid = threadIdx.x;
    float v = emb[i * 256 + tid];
    red[tid] = v * v;
    __syncthreads();
    for (int s = 128; s > 0; s >>= 1) {
        if (tid < s) red[tid] += red[tid + s];
        __syncthreads();
    }
    if (tid == 0) norms[i] = fmaxf(sqrtf(red[0]), 1e-12f);
}

// ---------------------------------------------------------------------------
// Kernel 4: cosine sims (128x128x256) + softmax(beta=8) + label aggregation.
// ---------------------------------------------------------------------------
__global__ void __launch_bounds__(128) attend_kernel(
        const float* __restrict__ emb, const float* __restrict__ norms,
        const int* __restrict__ labels, float* __restrict__ out) {
    __shared__ float qv[256];
    __shared__ float sim[128];
    __shared__ float red[128];
    int q = blockIdx.x, tid = threadIdx.x;
    qv[tid]       = emb[(128 + q) * 256 + tid];
    qv[128 + tid] = emb[(128 + q) * 256 + 128 + tid];
    __syncthreads();
    const float* se = emb + tid * 256;    // support rows 0..127
    float d = 0.f;
    for (int e = 0; e < 256; e++) d += qv[e] * se[e];
    float s = 8.f * d / (norms[128 + q] * norms[tid]);
    sim[tid] = s; red[tid] = s;
    __syncthreads();
    for (int st = 64; st > 0; st >>= 1) {
        if (tid < st) red[tid] = fmaxf(red[tid], red[tid + st]);
        __syncthreads();
    }
    float mx = red[0];
    __syncthreads();
    float e = expf(sim[tid] - mx);
    sim[tid] = e; red[tid] = e;
    __syncthreads();
    for (int st = 64; st > 0; st >>= 1) {
        if (tid < st) red[tid] += red[tid + st];
        __syncthreads();
    }
    float inv = 1.f / red[0];
    __syncthreads();
    if (tid < 10) {
        float acc = 0.f;
        for (int s2 = 0; s2 < 128; s2++)
            if (labels[s2] == tid) acc += sim[s2] * inv;
        out[q * 10 + tid] = acc;
    }
}

// ---------------------------------------------------------------------------
extern "C" void kernel_launch(void* const* d_in, const int* in_sizes, int n_in,
                              void* d_out, int out_size, void* d_ws, size_t ws_size,
                              hipStream_t stream) {
    (void)in_sizes; (void)n_in; (void)out_size; (void)ws_size;
    const float* sup    = (const float*)d_in[0];   // (16,128,1,32,32)
    const float* qry    = (const float*)d_in[1];   // (16,128,1,32,32)
    const float* w1     = (const float*)d_in[2];   // (64,1,5,5)
    const float* w2     = (const float*)d_in[3];   // (128,64,5,5)
    const float* projw  = (const float*)d_in[4];   // (256,8192)
    const int*   labels = (const int*)d_in[5];     // (128,)
    float* out = (float*)d_out;                    // (128,10)

    char* ws = (char*)d_ws;
    _Float16* bpack  = (_Float16*)(ws);                      // 409,600 B
    unsigned* p1bits = (unsigned*)(ws + 409600);             // 8,388,608 B
    float*    rate   = (float*)(ws + 409600 + 8388608);      // 8,388,608 B
    float*    emb    = (float*)(ws + 409600 + 16777216);     //   262,144 B
    float*    norms  = (float*)(ws + 409600 + 17039360);     //     1,024 B

    pack_w2<<<800, 256, 0, stream>>>(w2, bpack);
    lif1_kernel<<<256, 1024, 0, stream>>>(sup, qry, w1, p1bits);
    conv2_lif2_kernel<<<256, 256, SM_TOTAL, stream>>>(p1bits, bpack, rate);
    project_kernel<<<256, 256, 0, stream>>>(rate, projw, emb);
    rownorm_kernel<<<256, 256, 0, stream>>>(emb, norms);
    attend_kernel<<<128, 128, 0, stream>>>(emb, norms, labels, out);
}
